// GaussianMixtureMLP_64183991271937
// MI455X (gfx1250) — matrix operand order repaired and checked by reading, not compiled, not run on hardware
//
#include <hip/hip_runtime.h>

// ---- CDNA5 (gfx1250) wave32 WMMA ensemble-MLP kernel, v3 -------------------
// v3: biases hoisted out of the tile loops (were a per-tile global round-trip),
// B-fragments batch-loaded per tile so ds_load latency amortizes and WMMAs
// issue back-to-back, global_prefetch of the next weight block during staging.

typedef __attribute__((ext_vector_type(16))) __bf16 v16bf;
typedef __attribute__((ext_vector_type(8)))  __bf16 v8bf;
typedef __attribute__((ext_vector_type(2)))  __bf16 v2bf;
typedef __attribute__((ext_vector_type(8)))  float  v8f;

#define NM     5      // models
#define HID    128    // hidden = inputs
#define NOUTP  36     // 2*18 raw outputs per model
#define NOUT   18     // mixture outputs
#define WSTR   136    // padded LDS row stride (bf16 elems): bank shift 4/row

// A fragment from a global f32 row: elems 0..7 = cols col0..col0+7,
// elems 8..15 = cols col0+16..col0+23  (col0 = kbase + (lane>=16)*8).
static __device__ __forceinline__ v16bf afrag_g(const float* row, int col0) {
  const float4* q  = (const float4*)(row + col0);
  const float4* q2 = (const float4*)(row + col0 + 16);
  float4 a0 = q[0], a1 = q[1], a2 = q2[0], a3 = q2[1];
  v16bf a;
  a[0]=(__bf16)a0.x;  a[1]=(__bf16)a0.y;  a[2]=(__bf16)a0.z;  a[3]=(__bf16)a0.w;
  a[4]=(__bf16)a1.x;  a[5]=(__bf16)a1.y;  a[6]=(__bf16)a1.z;  a[7]=(__bf16)a1.w;
  a[8]=(__bf16)a2.x;  a[9]=(__bf16)a2.y;  a[10]=(__bf16)a2.z; a[11]=(__bf16)a2.w;
  a[12]=(__bf16)a3.x; a[13]=(__bf16)a3.y; a[14]=(__bf16)a3.z; a[15]=(__bf16)a3.w;
  return a;
}

// A fragment from the per-wave LDS h-buffer (bf16, row stride WSTR):
// two 16-byte ds_load_b128 reads (cols col0..col0+7 and col0+16..col0+23).
static __device__ __forceinline__ v16bf afrag_lds(const __bf16* row, int col0) {
  v8bf lo = *(const v8bf*)(row + col0);
  v8bf hi = *(const v8bf*)(row + col0 + 16);
  return __builtin_shufflevector(lo, hi, 0,1,2,3,4,5,6,7,8,9,10,11,12,13,14,15);
}

// B fragment from LDS weight buffer: bf16 of one W row at k0..k0+7 and
// k0+16..k0+23, k0 = kbase + (lane>=16)*8 -> two ds_load_b128.
static __device__ __forceinline__ v16bf bfrag_lds(const __bf16* row, int k0) {
  v8bf lo = *(const v8bf*)(row + k0);
  v8bf hi = *(const v8bf*)(row + k0 + 16);
  return __builtin_shufflevector(lo, hi, 0,1,2,3,4,5,6,7,8,9,10,11,12,13,14,15);
}

// Cooperative f32 -> bf16 weight staging: nrows rows of 128 cols into wbuf
// (row stride WSTR); rows >= validRows are zero-filled (layer-3 padding).
// While staging, stream prefetches for the NEXT weight block (global_prefetch).
static __device__ __forceinline__ void stage_weights(
    __bf16* wbuf, const float* Wsrc, int nrows, int validRows, int tid,
    const float* pfNext)
{
  const int total = nrows * 64;          // float2 pairs
  for (int p = tid; p < total; p += 256) {
    const int row = p >> 6;
    const int cp  = p & 63;
    v2bf d;
    if (row < validRows) {
      const float2 f = ((const float2*)(Wsrc + (size_t)row * HID))[cp];
      d[0] = (__bf16)f.x; d[1] = (__bf16)f.y;
    } else {
      d[0] = (__bf16)0.0f; d[1] = (__bf16)0.0f;
    }
    *(v2bf*)(wbuf + row * WSTR + cp * 2) = d;
    if (pfNext && ((p & 15) == 0))       // one prefetch per 128 B
      __builtin_prefetch(pfNext + (size_t)p * 2, 0, 1);
  }
}

__global__ __launch_bounds__(256) void ens_mlp_wmma(
    const float* __restrict__ x,
    const float* __restrict__ W1, const float* __restrict__ b1,
    const float* __restrict__ W2, const float* __restrict__ b2,
    const float* __restrict__ W3, const float* __restrict__ b3,
    float* __restrict__ out, int rows)
{
  // 64 W rows staged at a time (half of a 128-row layer); shared by all waves.
  __shared__ __align__(16) __bf16 wbuf[64 * WSTR];        // ~17 KB
  // Per-wave h re-layout buffer; its region is reused for the epilogue stage.
  __shared__ __align__(16) __bf16 hbuf[8][16 * WSTR];     // ~34 KB

  const int tid  = threadIdx.x;
  const int lane = tid & 31;
  const int wv   = tid >> 5;
  const int r    = lane & 15;     // A row / B,C,D column
  const int hi   = lane >> 4;     // lane-half selector
  const int row0 = (blockIdx.x * 8 + wv) * 16;

  // ---- x A-fragments: converted once, reused by all 5 models ----
  v16bf ax[4];
  {
    const float* xr = x + (size_t)(row0 + r) * HID;
#pragma unroll
    for (int c = 0; c < 4; ++c) ax[c] = afrag_g(xr, c * 32 + hi * 8);
  }

  float accA[3][8], accB[3][8];
#pragma unroll
  for (int t = 0; t < 3; ++t)
#pragma unroll
    for (int v = 0; v < 8; ++v) { accA[t][v] = 0.0f; accB[t][v] = 0.0f; }

  __bf16*       hrow0 = &hbuf[wv][0];
  const __bf16* hlane = &hbuf[wv][r * WSTR];

  for (int m = 0; m < NM; ++m) {
    const float* W1m = W1 + (size_t)m * HID * HID;
    const float* W2m = W2 + (size_t)m * HID * HID;
    const float* W3m = W3 + (size_t)m * NOUTP * HID;
    const float* b1m = b1 + m * HID;
    const float* b2m = b2 + m * HID;
    const float* b3m = b3 + m * NOUTP;
    const float* W1next = (m + 1 < NM) ? (W1 + (size_t)(m + 1) * HID * HID)
                                       : (const float*)0;

    // ---- hoisted per-lane bias values: issue early, overlap with staging ----
    float bv1[8], bv2[8], bv3[3];
#pragma unroll
    for (int nt = 0; nt < 8; ++nt) {
      bv1[nt] = b1m[nt * 16 + r];
      bv2[nt] = b2m[nt * 16 + r];
    }
#pragma unroll
    for (int nt = 0; nt < 3; ++nt) {
      const int nG = nt * 16 + r;
      bv3[nt] = (nG < NOUTP) ? b3m[nG] : 0.0f;
    }

    // ---------------- layer 1: h = relu(x @ W1^T + b1) ----------------
#pragma unroll
    for (int half = 0; half < 2; ++half) {
      __syncthreads();                                  // wbuf free
      stage_weights(wbuf, W1m + (size_t)half * 64 * HID, 64, 64, tid,
                    W2m + (size_t)half * 64 * HID);     // prefetch W2
      __syncthreads();                                  // wbuf ready
#pragma unroll
      for (int ntl = 0; ntl < 4; ++ntl) {
        const int nt = half * 4 + ntl;
        const __bf16* wrow = wbuf + (ntl * 16 + r) * WSTR;
        v16bf bb[4];
#pragma unroll
        for (int c = 0; c < 4; ++c) bb[c] = bfrag_lds(wrow, c * 32 + hi * 8);
        v8f acc = {0.f,0.f,0.f,0.f,0.f,0.f,0.f,0.f};
#pragma unroll
        for (int c = 0; c < 4; ++c)
          acc = __builtin_amdgcn_wmma_f32_16x16x32_bf16(
              false, ax[c], false, bb[c], (short)0, acc, false, false);
#pragma unroll
        for (int v = 0; v < 8; ++v) {
          float h = fmaxf(acc[v] + bv1[nt], 0.0f);
          hrow0[(v + hi * 8) * WSTR + (nt * 16 + r)] = (__bf16)h;
        }
      }
    }

    // ---------------- layer 2: h = relu(h @ W2^T + b2) ----------------
    v16bf ah[4];
#pragma unroll
    for (int c = 0; c < 4; ++c) ah[c] = afrag_lds(hlane, c * 32 + hi * 8);
#pragma unroll
    for (int half = 0; half < 2; ++half) {
      __syncthreads();
      stage_weights(wbuf, W2m + (size_t)half * 64 * HID, 64, 64, tid,
                    (half == 0) ? (W3m) : (const float*)0);  // prefetch W3
      __syncthreads();
#pragma unroll
      for (int ntl = 0; ntl < 4; ++ntl) {
        const int nt = half * 4 + ntl;
        const __bf16* wrow = wbuf + (ntl * 16 + r) * WSTR;
        v16bf bb[4];
#pragma unroll
        for (int c = 0; c < 4; ++c) bb[c] = bfrag_lds(wrow, c * 32 + hi * 8);
        v8f acc = {0.f,0.f,0.f,0.f,0.f,0.f,0.f,0.f};
#pragma unroll
        for (int c = 0; c < 4; ++c)
          acc = __builtin_amdgcn_wmma_f32_16x16x32_bf16(
              false, ah[c], false, bb[c], (short)0, acc, false, false);
#pragma unroll
        for (int v = 0; v < 8; ++v) {
          float h = fmaxf(acc[v] + bv2[nt], 0.0f);
          hrow0[(v + hi * 8) * WSTR + (nt * 16 + r)] = (__bf16)h;
        }
      }
    }

    // -------- layer 3 (36 rows staged + 12 zero-padded) + mixture acc --------
#pragma unroll
    for (int c = 0; c < 4; ++c) ah[c] = afrag_lds(hlane, c * 32 + hi * 8);
    __syncthreads();
    stage_weights(wbuf, W3m, 48, NOUTP, tid, W1next);   // prefetch next model
    __syncthreads();
#pragma unroll
    for (int nt = 0; nt < 3; ++nt) {
      const int nG = nt * 16 + r;
      const __bf16* wrow = wbuf + nG * WSTR;
      v16bf bb[4];
#pragma unroll
      for (int c = 0; c < 4; ++c) bb[c] = bfrag_lds(wrow, c * 32 + hi * 8);
      v8f acc = {0.f,0.f,0.f,0.f,0.f,0.f,0.f,0.f};
#pragma unroll
      for (int c = 0; c < 4; ++c)
        acc = __builtin_amdgcn_wmma_f32_16x16x32_bf16(
            false, ah[c], false, bb[c], (short)0, acc, false, false);
      const float bv = bv3[nt];
      if (nG < NOUT) {                 // mean column
#pragma unroll
        for (int v = 0; v < 8; ++v) {
          float mu = acc[v] + bv;
          accA[nt][v] += mu;
          accB[nt][v] += mu * mu;
        }
      } else if (nG < NOUTP) {         // variance column: softplus + 1e-6
#pragma unroll
        for (int v = 0; v < 8; ++v) {
          float t  = acc[v] + bv;
          float sp = fmaxf(t, 0.0f) + __logf(1.0f + __expf(-fabsf(t))) + 1e-6f;
          accA[nt][v] += sp;
        }
      }
    }
  }

  // ---- epilogue staging aliases this wave's (now dead) hbuf region ----
  asm volatile("" ::: "memory");       // forbid IR reordering across the reuse
  float* sA = (float*)hrow0;           // [16][36] : sum(mu) | sum(softplus)
  float* sB = sA + 16 * NOUTP;         // [16][18] : sum(mu^2)
#pragma unroll
  for (int nt = 0; nt < 3; ++nt) {
    const int col = nt * 16 + r;
#pragma unroll
    for (int v = 0; v < 8; ++v) {
      const int rr = v + hi * 8;
      if (col < NOUTP) sA[rr * NOUTP + col] = accA[nt][v];
      if (col < NOUT)  sB[rr * NOUT  + col] = accB[nt][v];
    }
  }

  // ---- final mixture combine + store (mean | variance), per-wave private ----
  const float inv = 1.0f / (float)NM;
  for (int idx = lane; idx < 16 * NOUT; idx += 32) {
    const int rr = idx / NOUT, o = idx % NOUT;
    float mu  = sA[rr * NOUTP + o] * inv;
    float ev  = (sA[rr * NOUTP + NOUT + o] + sB[rr * NOUT + o]) * inv - mu * mu;
    float var = fmaxf(ev, 0.0f) + 1e-6f;
    const size_t base = (size_t)(row0 + rr) * NOUT + o;
    out[base]                       = mu;
    out[(size_t)rows * NOUT + base] = var;
  }
}

extern "C" void kernel_launch(void* const* d_in, const int* in_sizes, int n_in,
                              void* d_out, int out_size, void* d_ws, size_t ws_size,
                              hipStream_t stream) {
  const float* x  = (const float*)d_in[0];
  const float* W1 = (const float*)d_in[1];
  const float* b1 = (const float*)d_in[2];
  const float* W2 = (const float*)d_in[3];
  const float* b2 = (const float*)d_in[4];
  const float* W3 = (const float*)d_in[5];
  const float* b3 = (const float*)d_in[6];
  float* out = (float*)d_out;

  const int rows   = in_sizes[0] / HID;   // 65536
  const int blocks = rows / (8 * 16);     // 8 waves/WG x 16 rows/wave

  ens_mlp_wmma<<<blocks, 256, 0, stream>>>(x, W1, b1, W2, b2, W3, b3, out, rows);
}
